// BLSTMLM_27547920237228
// MI455X (gfx1250) — compile-verified, matrix operand order checked by reading, not run on hardware
//
#include <hip/hip_runtime.h>

// ---------------------------------------------------------------------------
// BLSTM LM for MI455X (gfx1250, wave32).
// All matrix math via V_WMMA_F32_16X16X32_BF16 with bf16 hi/lo split (3 WMMAs
// per K-chunk) to preserve ~fp32 accuracy at bf16 matrix-pipe rates.
// A-tile staging uses GLOBAL_LOAD_ASYNC_TO_LDS_B128 (ASYNCcnt) when available.
// ---------------------------------------------------------------------------

typedef __bf16 v16bf __attribute__((ext_vector_type(16)));
typedef __bf16 v8bf  __attribute__((ext_vector_type(8)));
typedef float  v8f   __attribute__((ext_vector_type(8)));

#define B_   4
#define T_   512
#define V_   32000
#define E_   512
#define H_   512
#define G4H  2048   // 4*H
#define BT   2048   // B*T
#define H2   1024   // 2*H

#if defined(__gfx1250__) && __has_builtin(__builtin_amdgcn_global_load_async_to_lds_b128) && \
    __has_builtin(__builtin_amdgcn_s_wait_asynccnt)
#define USE_ASYNC_LDS 1
#define GLB1 __attribute__((address_space(1)))
#define LDS3 __attribute__((address_space(3)))
typedef int v4ig __attribute__((vector_size(16)));   // matches builtin param type
#else
#define USE_ASYNC_LDS 0
#endif

// ---- fp32 -> bf16 hi/lo split (round-to-nearest-even on both halves) ------
__device__ __forceinline__ void splitf(float x, unsigned short& hi, unsigned short& lo) {
    unsigned u  = __float_as_uint(x);
    unsigned r  = u + 0x7fffu + ((u >> 16) & 1u);
    hi = (unsigned short)(r >> 16);
    float fh = __uint_as_float(((unsigned)hi) << 16);
    float re = x - fh;
    unsigned u2 = __float_as_uint(re);
    unsigned r2 = u2 + 0x7fffu + ((u2 >> 16) & 1u);
    lo = (unsigned short)(r2 >> 16);
}

// ---- load one 16-element bf16 fragment half per ISA 16-bit A/B layout -----
// lane holds 8 elems at p[0..7] (K block) and 8 at p[16..23] (K block +16)
__device__ __forceinline__ v16bf load_frag(const unsigned short* p) {
    v8bf a = *(const v8bf*)p;
    v8bf b = *(const v8bf*)(p + 16);
    return __builtin_shufflevector(a, b, 0,1,2,3,4,5,6,7,8,9,10,11,12,13,14,15);
}

// ---- bf16x3 accumulate: acc += Ahi*Bhi + Ahi*Blo + Alo*Bhi ----------------
__device__ __forceinline__ v8f wmma3(v16bf ah, v16bf al, v16bf bh, v16bf bl, v8f acc) {
    acc = __builtin_amdgcn_wmma_f32_16x16x32_bf16(false, ah, false, bh, (short)0, acc, false, false);
    acc = __builtin_amdgcn_wmma_f32_16x16x32_bf16(false, ah, false, bl, (short)0, acc, false, false);
    acc = __builtin_amdgcn_wmma_f32_16x16x32_bf16(false, al, false, bh, (short)0, acc, false, false);
    return acc;
}

// ---------------------------------------------------------------------------
// Split a float array into bf16 hi/lo planes.
// ---------------------------------------------------------------------------
__global__ void k_split(const float* __restrict__ src, unsigned short* __restrict__ hi,
                        unsigned short* __restrict__ lo, int n) {
    int i = blockIdx.x * blockDim.x + threadIdx.x;
    if (i < n) { unsigned short h, l; splitf(src[i], h, l); hi[i] = h; lo[i] = l; }
}

// ---------------------------------------------------------------------------
// Embedding gather fused with bf16 split:  e[b,t,:] = emb[x[b,t],:]
// ---------------------------------------------------------------------------
__global__ void k_embed(const int* __restrict__ x, const float* __restrict__ emb,
                        unsigned short* __restrict__ ehi, unsigned short* __restrict__ elo) {
    int i = blockIdx.x * blockDim.x + threadIdx.x;      // over BT*E
    if (i >= BT * E_) return;
    int row = i >> 9, col = i & 511;
    int tok = x[row];
    float v = emb[(size_t)tok * E_ + col];
    unsigned short h, l; splitf(v, h, l);
    ehi[i] = h; elo[i] = l;
}

// ---------------------------------------------------------------------------
// Generic NT GEMM:  C[m,n] = sum_k A[m,k]*B[n,k] + bias[n]   (bf16x3, f32 acc)
// Block = 256 thr (8 waves, 2x4), block tile 32(M) x 64(N), K step 32.
// A staged in LDS (async-to-LDS when available), B per-lane from global.
// ---------------------------------------------------------------------------
#define LDK 40   // padded LDS row stride in ushorts (80B -> 20-bank row shift)

__global__ __launch_bounds__(256)
void k_gemm_bf16x3(const unsigned short* __restrict__ Ahi, const unsigned short* __restrict__ Alo,
                   const unsigned short* __restrict__ Bhi, const unsigned short* __restrict__ Blo,
                   const float* __restrict__ bias, float* __restrict__ C,
                   int M, int N, int K) {
    __shared__ unsigned short sAhi[32 * LDK];
    __shared__ unsigned short sAlo[32 * LDK];

    const int tid  = threadIdx.x;
    const int lane = tid & 31;
    const int w    = tid >> 5;      // wave 0..7
    const int wm   = w >> 2;        // 0..1  (M sub-tile)
    const int wn   = w & 3;         // 0..3  (N sub-tile)
    const int m0   = blockIdx.y * 32;
    const int n0   = blockIdx.x * 64;
    const int ln   = lane & 15;
    const int lh   = lane >> 4;     // lane half: K-block select

    const int arow = wm * 16 + ln;              // A row in LDS tile this lane reads
    const int bcol = n0 + wn * 16 + ln;         // output column n for this lane
    const unsigned short* pBh = Bhi + (size_t)bcol * K;
    const unsigned short* pBl = Blo + (size_t)bcol * K;

#if USE_ASYNC_LDS
    // async staging: 256 threads x one b128 -> 4KB tile (hi: tid<128, lo: tid>=128)
    const int srow = (tid & 127) >> 2;          // 0..31
    const int scol = (tid & 3) * 8;             // 0,8,16,24
    const unsigned short* gA   = ((tid < 128) ? Ahi : Alo) + (size_t)(m0 + srow) * K + scol;
    unsigned short*       lA   = ((tid < 128) ? sAhi : sAlo) + srow * LDK + scol;
    LDS3 v4ig* lA3 = (LDS3 v4ig*)(unsigned int)(unsigned long long)lA;
#else
    const int crow = tid >> 3;                  // coop-load: 32x32 elems, 4/thread
    const int ccol = (tid & 7) * 4;
#endif

    v8f acc = {};
    for (int k0 = 0; k0 < K; k0 += 32) {
#if USE_ASYNC_LDS
        __builtin_amdgcn_global_load_async_to_lds_b128(
            (GLB1 v4ig*)(unsigned long long)(gA + k0), lA3, 0, 0);
        __builtin_amdgcn_s_wait_asynccnt(0);
        __syncthreads();
#else
        const size_t goff = (size_t)(m0 + crow) * K + k0 + ccol;
        *(unsigned long long*)&sAhi[crow * LDK + ccol] = *(const unsigned long long*)(Ahi + goff);
        *(unsigned long long*)&sAlo[crow * LDK + ccol] = *(const unsigned long long*)(Alo + goff);
        __syncthreads();
#endif
        if (k0 + 32 < K) {                       // warm next B chunk in GL2
            __builtin_prefetch(pBh + k0 + 32, 0, 1);
            __builtin_prefetch(pBl + k0 + 32, 0, 1);
        }

        v16bf afh = load_frag(&sAhi[arow * LDK + lh * 8]);
        v16bf afl = load_frag(&sAlo[arow * LDK + lh * 8]);
        v16bf bfh = load_frag(pBh + k0 + lh * 8);
        v16bf bfl = load_frag(pBl + k0 + lh * 8);
        acc = wmma3(afh, afl, bfh, bfl, acc);
        __syncthreads();
    }

    // epilogue: C layout -> lane ln = n, VGPR r -> m = r + 8*lh
    const float bv = bias ? bias[bcol] : 0.f;
    const int mbase = m0 + wm * 16 + lh * 8;
    #pragma unroll
    for (int r = 0; r < 8; ++r)
        C[(size_t)(mbase + r) * N + bcol] = acc[r] + bv;
}

// ---------------------------------------------------------------------------
// Persistent bidirectional LSTM recurrence. grid.x = 2 (dir), 1024 thr/blk.
// h kept as bf16 hi/lo in LDS (16 rows, rows 4..15 zero), c in registers,
// gate pre-activations staged in LDS. w_hh (bf16 hi/lo, 4MB/dir) lives in L2.
// A fragments hoisted across the 4 N-tiles each wave owns.
// ---------------------------------------------------------------------------
#define HSTR 520   // padded LDS h-row stride in ushorts (1040B, 16B aligned)

__global__ __launch_bounds__(1024)
void k_lstm(const float* __restrict__ gx_f, const float* __restrict__ gx_b,
            const unsigned short* __restrict__ whhf_hi, const unsigned short* __restrict__ whhf_lo,
            const unsigned short* __restrict__ whhb_hi, const unsigned short* __restrict__ whhb_lo,
            float* __restrict__ hcat) {
    extern __shared__ char smem[];
    unsigned short* h_hi = (unsigned short*)smem;             // 16*HSTR
    unsigned short* h_lo = h_hi + 16 * HSTR;                  // 16*HSTR
    float*          g_s  = (float*)(h_lo + 16 * HSTR);        // 4*G4H

    const int dir = blockIdx.x;
    const float* gxd = dir ? gx_b : gx_f;
    const unsigned short* whh_hi = dir ? whhb_hi : whhf_hi;
    const unsigned short* whh_lo = dir ? whhb_lo : whhf_lo;

    const int tid  = threadIdx.x;
    const int lane = tid & 31;
    const int wid  = tid >> 5;            // 0..31
    const int ln   = lane & 15;
    const int lh   = lane >> 4;

    // zero h state (all 16 rows; rows 4..15 stay zero forever)
    for (int i = tid; i < 16 * HSTR; i += 1024) { h_hi[i] = 0; h_lo[i] = 0; }
    float c0 = 0.f, c1 = 0.f;
    // each thread owns two cell states: s and s+1024  -> (b, j)
    const int b0i = tid >> 9,          j0 = tid & 511;          // b 0..1
    const int b1i = (tid + 1024) >> 9, j1 = tid & 511;          // b 2..3

    // per-wave w_hh row pointers for the 4 owned N-tiles
    const unsigned short* pwh[4];
    const unsigned short* pwl[4];
    #pragma unroll
    for (int q = 0; q < 4; ++q) {
        const int ncol = wid * 64 + q * 16 + ln;
        pwh[q] = whh_hi + (size_t)ncol * H_;
        pwl[q] = whh_lo + (size_t)ncol * H_;
    }
    __syncthreads();

    for (int step = 0; step < T_; ++step) {
        const int t = dir ? (T_ - 1 - step) : step;

        // ---- g = h @ w_hh^T  (M=16 padded, N=2048, K=512); A frags shared
        v8f acc[4] = {};
        for (int k0 = 0; k0 < H_; k0 += 32) {
            v16bf afh = load_frag(&h_hi[ln * HSTR + k0 + lh * 8]);
            v16bf afl = load_frag(&h_lo[ln * HSTR + k0 + lh * 8]);
            #pragma unroll
            for (int q = 0; q < 4; ++q) {
                v16bf bfh = load_frag(pwh[q] + k0 + lh * 8);
                v16bf bfl = load_frag(pwl[q] + k0 + lh * 8);
                acc[q] = wmma3(afh, afl, bfh, bfl, acc[q]);
            }
        }
        if (lh == 0) {                 // rows M=0..3 valid (VGPR r = M)
            #pragma unroll
            for (int q = 0; q < 4; ++q) {
                const int n = wid * 64 + q * 16 + ln;
                g_s[0 * G4H + n] = acc[q][0];
                g_s[1 * G4H + n] = acc[q][1];
                g_s[2 * G4H + n] = acc[q][2];
                g_s[3 * G4H + n] = acc[q][3];
            }
        }
        __syncthreads();

        // ---- gates + state update; 2 cells per thread
        {
            const int m = b0i * T_ + t;
            const float* gr = gxd + (size_t)m * G4H;
            const float* gs = g_s + b0i * G4H;
            float xi = gr[j0]          + gs[j0];
            float xf = gr[512  + j0]   + gs[512  + j0];
            float xo = gr[1024 + j0]   + gs[1024 + j0];
            float xu = gr[1536 + j0]   + gs[1536 + j0];
            float ig = 1.f / (1.f + __expf(-xi));
            float fg = 1.f / (1.f + __expf(-xf));
            float og = 1.f / (1.f + __expf(-xo));
            float ug = tanhf(xu);
            c0 = fg * c0 + ig * ug;
            float hv = og * tanhf(c0);
            hcat[(size_t)m * H2 + dir * H_ + j0] = hv;
            unsigned short hh, hl; splitf(hv, hh, hl);
            h_hi[b0i * HSTR + j0] = hh; h_lo[b0i * HSTR + j0] = hl;
        }
        {
            const int m = b1i * T_ + t;
            const float* gr = gxd + (size_t)m * G4H;
            const float* gs = g_s + b1i * G4H;
            float xi = gr[j1]          + gs[j1];
            float xf = gr[512  + j1]   + gs[512  + j1];
            float xo = gr[1024 + j1]   + gs[1024 + j1];
            float xu = gr[1536 + j1]   + gs[1536 + j1];
            float ig = 1.f / (1.f + __expf(-xi));
            float fg = 1.f / (1.f + __expf(-xf));
            float og = 1.f / (1.f + __expf(-xo));
            float ug = tanhf(xu);
            c1 = fg * c1 + ig * ug;
            float hv = og * tanhf(c1);
            hcat[(size_t)m * H2 + dir * H_ + j1] = hv;
            unsigned short hh, hl; splitf(hv, hh, hl);
            h_hi[b1i * HSTR + j1] = hh; h_lo[b1i * HSTR + j1] = hl;
        }
        __syncthreads();
    }
}

// ---------------------------------------------------------------------------
extern "C" void kernel_launch(void* const* d_in, const int* in_sizes, int n_in,
                              void* d_out, int out_size, void* d_ws, size_t ws_size,
                              hipStream_t stream) {
    const int*   x      = (const int*)  d_in[0];
    const float* emb    = (const float*)d_in[1];
    const float* w_ih_f = (const float*)d_in[2];
    const float* b_ih_f = (const float*)d_in[3];
    const float* w_hh_f = (const float*)d_in[4];
    const float* w_ih_b = (const float*)d_in[5];
    const float* b_ih_b = (const float*)d_in[6];
    const float* w_hh_b = (const float*)d_in[7];
    const float* proj_w = (const float*)d_in[8];
    const float* proj_b = (const float*)d_in[9];
    float* out = (float*)d_out;

    size_t off = 0;
    auto carve = [&](size_t bytes) -> char* {
        char* p = (char*)d_ws + off;
        off += (bytes + 255) & ~(size_t)255;
        return p;
    };
    unsigned short* ehi     = (unsigned short*)carve((size_t)BT * E_ * 2);
    unsigned short* elo     = (unsigned short*)carve((size_t)BT * E_ * 2);
    unsigned short* wihf_hi = (unsigned short*)carve((size_t)G4H * E_ * 2);
    unsigned short* wihf_lo = (unsigned short*)carve((size_t)G4H * E_ * 2);
    unsigned short* wihb_hi = (unsigned short*)carve((size_t)G4H * E_ * 2);
    unsigned short* wihb_lo = (unsigned short*)carve((size_t)G4H * E_ * 2);
    unsigned short* whhf_hi = (unsigned short*)carve((size_t)G4H * H_ * 2);
    unsigned short* whhf_lo = (unsigned short*)carve((size_t)G4H * H_ * 2);
    unsigned short* whhb_hi = (unsigned short*)carve((size_t)G4H * H_ * 2);
    unsigned short* whhb_lo = (unsigned short*)carve((size_t)G4H * H_ * 2);
    unsigned short* pw_hi   = (unsigned short*)carve((size_t)V_ * H2 * 2);
    unsigned short* pw_lo   = (unsigned short*)carve((size_t)V_ * H2 * 2);
    float*          gx_f    = (float*)carve((size_t)BT * G4H * 4);
    float*          gx_b    = (float*)carve((size_t)BT * G4H * 4);
    float*          hcat    = (float*)carve((size_t)BT * H2 * 4);
    unsigned short* hc_hi   = (unsigned short*)carve((size_t)BT * H2 * 2);
    unsigned short* hc_lo   = (unsigned short*)carve((size_t)BT * H2 * 2);

    const int nW = G4H * E_;            // 2048*512
    k_split<<<(nW + 255) / 256, 256, 0, stream>>>(w_ih_f, wihf_hi, wihf_lo, nW);
    k_split<<<(nW + 255) / 256, 256, 0, stream>>>(w_ih_b, wihb_hi, wihb_lo, nW);
    k_split<<<(nW + 255) / 256, 256, 0, stream>>>(w_hh_f, whhf_hi, whhf_lo, nW);
    k_split<<<(nW + 255) / 256, 256, 0, stream>>>(w_hh_b, whhb_hi, whhb_lo, nW);
    const int nP = V_ * H2;             // 32000*1024
    k_split<<<(nP + 255) / 256, 256, 0, stream>>>(proj_w, pw_hi, pw_lo, nP);

    k_embed<<<(BT * E_ + 255) / 256, 256, 0, stream>>>(x, emb, ehi, elo);

    // input projections: gx = e @ w_ih^T + b_ih   (M=2048, N=2048, K=512)
    dim3 gin(G4H / 64, BT / 32);
    k_gemm_bf16x3<<<gin, 256, 0, stream>>>(ehi, elo, wihf_hi, wihf_lo, b_ih_f, gx_f, BT, G4H, E_);
    k_gemm_bf16x3<<<gin, 256, 0, stream>>>(ehi, elo, wihb_hi, wihb_lo, b_ih_b, gx_b, BT, G4H, E_);

    // recurrence: one persistent 32-wave workgroup per direction
    const size_t lds = (size_t)(2 * 16 * HSTR) * 2 + (size_t)(4 * G4H) * 4;   // 66048 B
    k_lstm<<<2, 1024, lds, stream>>>(gx_f, gx_b, whhf_hi, whhf_lo, whhb_hi, whhb_lo, hcat);

    k_split<<<(BT * H2 + 255) / 256, 256, 0, stream>>>(hcat, hc_hi, hc_lo, BT * H2);

    // output projection: out = hcat @ proj_w^T + proj_b  (M=2048, N=32000, K=1024)
    dim3 gout(V_ / 64, BT / 32);
    k_gemm_bf16x3<<<gout, 256, 0, stream>>>(hc_hi, hc_lo, pw_hi, pw_lo, proj_b, out, BT, V_, H2);

    (void)in_sizes; (void)n_in; (void)out_size; (void)ws_size;
}
